// OutputBlock_67783173865711
// MI455X (gfx1250) — matrix operand order repaired and checked by reading, not compiled, not run on hardware
//
#include <hip/hip_runtime.h>
#include <hip/hip_bf16.h>

// Sizes from the reference
#define NA_  20000
#define NE_  240000
#define NT_  300000
#define DA_  256
#define DE_  256
#define DR_  16
#define DC_  16
#define NH_  3
#define NV_  9
#define HID_ 64

typedef __attribute__((ext_vector_type(16))) __bf16 v16bf;
typedef __attribute__((ext_vector_type(8)))  float  v8f;
typedef __attribute__((ext_vector_type(4)))  unsigned int u32x4;

union Frag {
    v16bf v;
    unsigned short u[16];
    u32x4 q[2];
};

__device__ inline unsigned short f2bf(float x) {
    unsigned int u = __float_as_uint(x);
    u += 0x7fffu + ((u >> 16) & 1u);   // round-to-nearest-even
    return (unsigned short)(u >> 16);
}
__device__ inline float bf2f(unsigned short b) {
    return __uint_as_float(((unsigned int)b) << 16);
}
__device__ inline unsigned int pk2bf(float a, float b) {
    return (unsigned int)f2bf(a) | ((unsigned int)f2bf(b) << 16);
}
__device__ inline float silu_f(float x)  { return x / (1.0f + __expf(-x)); }
__device__ inline float ssilu_f(float x) { return (x / (1.0f + __expf(-x))) * (1.0f / 0.6f); }

__device__ inline v8f v8zero() {
    v8f z;
#pragma unroll
    for (int i = 0; i < 8; ++i) z[i] = 0.0f;
    return z;
}

// A fragment: 16xK tile in LDS, row-major with given stride (elements).
// lane = (row = lane&15, khalf = lane>>4); elems 0..7 = K kb+khalf*8+0..7,
// elems 8..15 = K kb+16+khalf*8+0..7  (ISA 7.12.2, 16-bit A 16x32 layout).
__device__ inline v16bf load_a(const unsigned short* lds, int row, int kb, int khalf, int stride) {
    Frag f;
    const unsigned short* p = lds + row * stride + kb + khalf * 8;
    f.q[0] = *(const u32x4*)p;
    f.q[1] = *(const u32x4*)(p + 16);
    return f.v;
}
// B fragment: g = &W[kb+khalf*8][nb+(lane&15)]; elems 0..7 = K +0..7, elems 8..15 = K +16..23 (stride ld).
__device__ inline v16bf load_b(const unsigned short* g, int ld) {
    Frag f;
#pragma unroll
    for (int s = 0; s < 8; ++s) {
        f.u[s] = g[s * ld];
        f.u[8 + s] = g[(16 + s) * ld];
    }
    return f.v;
}
__device__ inline v8f wmma_bf16(v16bf a, v16bf b, v8f c) {
    return __builtin_amdgcn_wmma_f32_16x16x32_bf16(false, a, false, b, (short)0, c, false, false);
}

// ---------------- prep kernels ----------------
__global__ void zero_f32_kernel(float* __restrict__ p, int n) {
    int i = blockIdx.x * 256 + threadIdx.x;
    if (i < n) p[i] = 0.0f;
}
__global__ void cvt_bf16_kernel(const float* __restrict__ src, unsigned short* __restrict__ dst,
                                int n, int ntot) {
    int i = blockIdx.x * 256 + threadIdx.x;
    if (i < ntot) dst[i] = (i < n) ? f2bf(src[i]) : (unsigned short)0;
}
// Ws3 (64,9) -> (64,16) with N padded by zeros
__global__ void cvt_ws3_kernel(const float* __restrict__ src, unsigned short* __restrict__ dst) {
    int i = blockIdx.x * 256 + threadIdx.x;
    if (i < 64 * 16) {
        int r = i >> 4, c = i & 15;
        dst[i] = (c < NV_) ? f2bf(src[r * NV_ + c]) : (unsigned short)0;
    }
}

// ---------------- energy scatter: xE[idj] += m * (rbf @ W_rbf_E) ----------------
__global__ __launch_bounds__(256) void energy_scatter_kernel(
    const float* __restrict__ m, const float* __restrict__ rbf,
    const float* __restrict__ WrbfE, const int* __restrict__ idj,
    float* __restrict__ xE, int nE)
{
    int e = blockIdx.x;
    int t = threadIdx.x;
    if (e >= nE) return;
    const float* rr = rbf + (size_t)e * DR_;
    float g = 0.0f;
#pragma unroll
    for (int k = 0; k < DR_; ++k) g += rr[k] * WrbfE[k * DE_ + t];
    float v = m[(size_t)e * DE_ + t] * g;
    int a = idj[e];
    unsafeAtomicAdd(&xE[(size_t)a * DE_ + t], v);
}

// ---------------- fused MLP (+optional rbf gate) + 256->1 projection ----------------
// One wave owns 16 rows; 7 GEMMs (16x256)@(256x256) via bf16 WMMA, f32 accum.
__device__ inline void gemm16x256(const unsigned short* lds, const unsigned short* __restrict__ W,
                                  int lane, v8f acc[16]) {
    const int row = lane & 15, khalf = lane >> 4;
#pragma unroll
    for (int nt = 0; nt < 16; ++nt) acc[nt] = v8zero();
    for (int kb = 0; kb < 256; kb += 32) {
        v16bf a = load_a(lds, row, kb, khalf, 256);
        const unsigned short* wp = W + (size_t)(kb + khalf * 8) * 256 + row;
#pragma unroll
        for (int nt = 0; nt < 16; ++nt) {
            v16bf b = load_b(wp + nt * 16, 256);
            acc[nt] = wmma_bf16(a, b, acc[nt]);
        }
    }
}

__global__ __launch_bounds__(256) void mlp_out_kernel(
    const float* __restrict__ X,              // (nRows,256) f32
    const unsigned short* __restrict__ Wd1,   // 256x256 bf16
    const unsigned short* __restrict__ Wres,  // 6 x 256x256 bf16
    const float* __restrict__ Wout,           // 256 f32
    const float* __restrict__ rbf,            // (nRows,16) f32 or nullptr
    const float* __restrict__ Wgate,          // 16x256 f32 or nullptr
    float* __restrict__ out,                  // (nRows) f32
    int nRows)
{
    __shared__ __align__(16) unsigned short XA[8][16 * 256];
    __shared__ __align__(16) unsigned short XB[8][16 * 256];
    __shared__ __align__(16) float WG[16 * 256];

    const int tid = threadIdx.x, wave = tid >> 5, lane = tid & 31;
    const int khalf = lane >> 4, n0 = lane & 15;
    const int rowBase = blockIdx.x * 128 + wave * 16;
    unsigned short* A = XA[wave];
    unsigned short* B = XB[wave];
    const bool gated = (rbf != nullptr);

    if (gated) {   // block-uniform branch
        for (int i = tid; i < 16 * 256; i += 256) WG[i] = Wgate[i];
        __syncthreads();
    }

    // Load + convert the 16x256 input tile (zeros past nRows); packed 16B LDS stores
    for (int rr = 0; rr < 16; ++rr) {
        int gr = rowBase + rr;
        int c = lane * 8;
        float4 v0 = make_float4(0.f, 0.f, 0.f, 0.f), v1 = v0;
        if (gr < nRows) {
            const float4* p = (const float4*)(X + (size_t)gr * 256 + c);
            v0 = p[0]; v1 = p[1];
        }
        u32x4 pk;
        pk[0] = pk2bf(v0.x, v0.y); pk[1] = pk2bf(v0.z, v0.w);
        pk[2] = pk2bf(v1.x, v1.y); pk[3] = pk2bf(v1.z, v1.w);
        *(u32x4*)(A + rr * 256 + c) = pk;
    }

    v8f acc[16];

    // x = ssilu(x0 @ Wd1)
    gemm16x256(A, Wd1, lane, acc);
#pragma unroll
    for (int nt = 0; nt < 16; ++nt)
#pragma unroll
        for (int r = 0; r < 8; ++r)
            A[(r + 8 * khalf) * 256 + nt * 16 + n0] = f2bf(ssilu_f(acc[nt][r]));

    // 3 residual blocks
    for (int i = 0; i < NH_; ++i) {
        gemm16x256(A, Wres + (size_t)(2 * i) * 65536, lane, acc);
#pragma unroll
        for (int nt = 0; nt < 16; ++nt)
#pragma unroll
            for (int r = 0; r < 8; ++r)
                B[(r + 8 * khalf) * 256 + nt * 16 + n0] = f2bf(ssilu_f(acc[nt][r]));

        gemm16x256(B, Wres + (size_t)(2 * i + 1) * 65536, lane, acc);
#pragma unroll
        for (int nt = 0; nt < 16; ++nt)
#pragma unroll
            for (int r = 0; r < 8; ++r) {
                int mrow = r + 8 * khalf, n = nt * 16 + n0;
                float z = ssilu_f(acc[nt][r]);
                float xo = bf2f(A[mrow * 256 + n]);
                float xn = (xo + z) * 0.70710678118654752440f;
                acc[nt][r] = xn;
                A[mrow * 256 + n] = f2bf(xn);
            }
    }

    // Optional gate (x * (rbf@Wgate)) then dot with Wout, reduce across 16 lanes
    float partial[8];
#pragma unroll
    for (int r = 0; r < 8; ++r) partial[r] = 0.0f;
#pragma unroll
    for (int r = 0; r < 8; ++r) {
        int mrow = r + 8 * khalf;
        float rb[16];
        if (gated) {
            int gr = rowBase + mrow;
            const float* rp = rbf + (size_t)(gr < nRows ? gr : 0) * DR_;
#pragma unroll
            for (int k = 0; k < 16; ++k) rb[k] = rp[k];
        }
#pragma unroll
        for (int nt = 0; nt < 16; ++nt) {
            int n = nt * 16 + n0;
            float v = acc[nt][r];
            if (gated) {
                float g = 0.0f;
#pragma unroll
                for (int k = 0; k < 16; ++k) g += rb[k] * WG[k * 256 + n];
                v *= g;
            }
            partial[r] += v * Wout[n];
        }
    }
#pragma unroll
    for (int off = 1; off < 16; off <<= 1)
#pragma unroll
        for (int r = 0; r < 8; ++r) partial[r] += __shfl_xor(partial[r], off, 32);
    if (n0 == 0) {
#pragma unroll
        for (int r = 0; r < 8; ++r) {
            int gr = rowBase + r + 8 * khalf;
            if (gr < nRows) out[gr] = partial[r];
        }
    }
}

// ---------------- stress branch: gather + 560->64->64->9 MLP ----------------
__global__ __launch_bounds__(256) void stress_kernel(
    const float* __restrict__ m, const float* __restrict__ rbf, const float* __restrict__ cbf,
    const int* __restrict__ id3i, const int* __restrict__ id3j,
    const unsigned short* __restrict__ Ws1p,  // 576x64 bf16 (K padded)
    const unsigned short* __restrict__ Ws2b,  // 64x64 bf16
    const unsigned short* __restrict__ Ws3p,  // 64x16 bf16 (N padded)
    float* __restrict__ out, int nT)
{
    __shared__ __align__(16) unsigned short EMB[8][16 * 576];
    __shared__ __align__(16) unsigned short HS[8][16 * 64];

    const int tid = threadIdx.x, wave = tid >> 5, lane = tid & 31;
    const int tBase = blockIdx.x * 128 + wave * 16;
    unsigned short* E = EMB[wave];
    unsigned short* H = HS[wave];

    // Gather 16 triplet rows: [m_i | m_j | rbf_i | rbf_j | cbf | pad16]
    // Vectorized: each lane covers 8 cols (2x float4 loads -> 1x 16B LDS store).
    for (int rr = 0; rr < 16; ++rr) {
        int t = tBase + rr;
        unsigned short* er = E + rr * 576;
        int c = lane * 8;
        if (t < nT) {
            int i = id3i[t], j = id3j[t];
            const float* mi = m + (size_t)i * DE_;
            const float* mj = m + (size_t)j * DE_;
            const float4* pi = (const float4*)(mi + c);
            const float4* pj = (const float4*)(mj + c);
            float4 a0 = pi[0], a1 = pi[1];
            float4 b0 = pj[0], b1 = pj[1];
            u32x4 w;
            w[0] = pk2bf(a0.x, a0.y); w[1] = pk2bf(a0.z, a0.w);
            w[2] = pk2bf(a1.x, a1.y); w[3] = pk2bf(a1.z, a1.w);
            *(u32x4*)(er + c) = w;
            w[0] = pk2bf(b0.x, b0.y); w[1] = pk2bf(b0.z, b0.w);
            w[2] = pk2bf(b1.x, b1.y); w[3] = pk2bf(b1.z, b1.w);
            *(u32x4*)(er + 256 + c) = w;
            if (lane < 16) {
                er[512 + lane] = f2bf(rbf[(size_t)i * DR_ + lane]);
                er[528 + lane] = f2bf(rbf[(size_t)j * DR_ + lane]);
                er[544 + lane] = f2bf(cbf[(size_t)t * DC_ + lane]);
                er[560 + lane] = 0;
            }
        } else {
            u32x4 z; z[0] = 0u; z[1] = 0u; z[2] = 0u; z[3] = 0u;
            *(u32x4*)(er + c) = z;
            *(u32x4*)(er + 256 + c) = z;
            if (lane < 16) {
                er[512 + lane] = 0; er[528 + lane] = 0;
                er[544 + lane] = 0; er[560 + lane] = 0;
            }
        }
    }

    const int row = lane & 15, khalf = lane >> 4, n0 = lane & 15;

    // GEMM1: (16x576)@(576x64)
    v8f acc[4];
#pragma unroll
    for (int nt = 0; nt < 4; ++nt) acc[nt] = v8zero();
    for (int kb = 0; kb < 576; kb += 32) {
        v16bf a = load_a(E, row, kb, khalf, 576);
#pragma unroll
        for (int nt = 0; nt < 4; ++nt) {
            v16bf b = load_b(Ws1p + (size_t)(kb + khalf * 8) * 64 + nt * 16 + n0, 64);
            acc[nt] = wmma_bf16(a, b, acc[nt]);
        }
    }
#pragma unroll
    for (int nt = 0; nt < 4; ++nt)
#pragma unroll
        for (int r = 0; r < 8; ++r)
            H[(r + 8 * khalf) * 64 + nt * 16 + n0] = f2bf(silu_f(acc[nt][r]));

    // GEMM2: (16x64)@(64x64)
#pragma unroll
    for (int nt = 0; nt < 4; ++nt) acc[nt] = v8zero();
    for (int kb = 0; kb < 64; kb += 32) {
        v16bf a = load_a(H, row, kb, khalf, 64);
#pragma unroll
        for (int nt = 0; nt < 4; ++nt) {
            v16bf b = load_b(Ws2b + (size_t)(kb + khalf * 8) * 64 + nt * 16 + n0, 64);
            acc[nt] = wmma_bf16(a, b, acc[nt]);
        }
    }
#pragma unroll
    for (int nt = 0; nt < 4; ++nt)
#pragma unroll
        for (int r = 0; r < 8; ++r)
            H[(r + 8 * khalf) * 64 + nt * 16 + n0] = f2bf(silu_f(acc[nt][r]));

    // GEMM3: (16x64)@(64x16), N padded; keep n<9
    v8f a3 = v8zero();
    for (int kb = 0; kb < 64; kb += 32) {
        v16bf a = load_a(H, row, kb, khalf, 64);
        v16bf b = load_b(Ws3p + (size_t)(kb + khalf * 8) * 16 + n0, 16);
        a3 = wmma_bf16(a, b, a3);
    }
    if (n0 < NV_) {
#pragma unroll
        for (int r = 0; r < 8; ++r) {
            int t = tBase + r + 8 * khalf;
            if (t < nT) out[(size_t)t * NV_ + n0] = a3[r];
        }
    }
}

// ---------------- launch ----------------
extern "C" void kernel_launch(void* const* d_in, const int* in_sizes, int n_in,
                              void* d_out, int out_size, void* d_ws, size_t ws_size,
                              hipStream_t stream) {
    (void)in_sizes; (void)n_in; (void)out_size; (void)ws_size;
    const float* m       = (const float*)d_in[1];
    const float* rbf     = (const float*)d_in[2];
    const float* cbf     = (const float*)d_in[3];
    const int*   id_j    = (const int*)d_in[4];
    const int*   id3_i   = (const int*)d_in[5];
    const int*   id3_j   = (const int*)d_in[6];
    const float* W_rbf_E = (const float*)d_in[7];
    const float* We_d1   = (const float*)d_in[8];
    const float* We_res  = (const float*)d_in[9];
    const float* W_out_E = (const float*)d_in[10];
    const float* Wf_d1   = (const float*)d_in[11];
    const float* Wf_res  = (const float*)d_in[12];
    const float* W_rbf_F = (const float*)d_in[13];
    const float* W_out_F = (const float*)d_in[14];
    const float* Ws1     = (const float*)d_in[15];
    const float* Ws2     = (const float*)d_in[16];
    const float* Ws3     = (const float*)d_in[17];

    char* ws = (char*)d_ws;
    size_t off = 0;
    auto alloc = [&](size_t bytes) -> void* {
        off = (off + 255) & ~(size_t)255;
        void* p = ws + off;
        off += bytes;
        return p;
    };

    float* xE = (float*)alloc((size_t)NA_ * DE_ * sizeof(float));
    unsigned short* bWe_d1  = (unsigned short*)alloc((size_t)65536 * 2);
    unsigned short* bWe_res = (unsigned short*)alloc((size_t)6 * 65536 * 2);
    unsigned short* bWf_d1  = (unsigned short*)alloc((size_t)65536 * 2);
    unsigned short* bWf_res = (unsigned short*)alloc((size_t)6 * 65536 * 2);
    unsigned short* bWs1    = (unsigned short*)alloc((size_t)576 * 64 * 2);
    unsigned short* bWs2    = (unsigned short*)alloc((size_t)64 * 64 * 2);
    unsigned short* bWs3    = (unsigned short*)alloc((size_t)64 * 16 * 2);

    // weight conversion
    {
        int n = 65536;
        cvt_bf16_kernel<<<(n + 255) / 256, 256, 0, stream>>>(We_d1, bWe_d1, n, n);
        cvt_bf16_kernel<<<(n + 255) / 256, 256, 0, stream>>>(Wf_d1, bWf_d1, n, n);
        int n6 = 6 * 65536;
        cvt_bf16_kernel<<<(n6 + 255) / 256, 256, 0, stream>>>(We_res, bWe_res, n6, n6);
        cvt_bf16_kernel<<<(n6 + 255) / 256, 256, 0, stream>>>(Wf_res, bWf_res, n6, n6);
        int np = 576 * 64;
        cvt_bf16_kernel<<<(np + 255) / 256, 256, 0, stream>>>(Ws1, bWs1, 560 * 64, np);
        cvt_bf16_kernel<<<(4096 + 255) / 256, 256, 0, stream>>>(Ws2, bWs2, 4096, 4096);
        cvt_ws3_kernel<<<4, 256, 0, stream>>>(Ws3, bWs3);
    }

    // energy accumulate
    zero_f32_kernel<<<(NA_ * DE_ + 255) / 256, 256, 0, stream>>>(xE, NA_ * DE_);
    energy_scatter_kernel<<<NE_, 256, 0, stream>>>(m, rbf, W_rbf_E, id_j, xE, NE_);

    float* outE = (float*)d_out;
    float* outF = outE + NA_;
    float* outS = outF + NE_;

    mlp_out_kernel<<<(NA_ + 127) / 128, 256, 0, stream>>>(
        xE, bWe_d1, bWe_res, W_out_E, nullptr, nullptr, outE, NA_);
    mlp_out_kernel<<<(NE_ + 127) / 128, 256, 0, stream>>>(
        m, bWf_d1, bWf_res, W_out_F, rbf, W_rbf_F, outF, NE_);
    stress_kernel<<<(NT_ + 127) / 128, 256, 0, stream>>>(
        m, rbf, cbf, id3_i, id3_j, bWs1, bWs2, bWs3, outS, NT_);
}
